// Attend_11519102288114
// MI455X (gfx1250) — matrix operand order repaired and checked by reading, not compiled, not run on hardware
//
#include <hip/hip_runtime.h>
#include <hip/hip_bf16.h>

typedef __attribute__((ext_vector_type(16))) __bf16 v16bf;
typedef __attribute__((ext_vector_type(8)))  __bf16 v8bf;
typedef __attribute__((ext_vector_type(2)))  __bf16 v2bf;
typedef __attribute__((ext_vector_type(8)))  float  v8f;

#define B_  4
#define H_  16
#define N_  2048
#define D_  64
#define BM  128       // query rows per workgroup
#define BN  64        // kv rows per iteration
#define NWAVES 8
#define NTHREADS 256
#define NTILES (N_ / BN)
#define SCALEF 0.125f            // 1/sqrt(64)
#define LOG2E  1.4426950408889634f

// ---- packed fp32x2 -> bf16x2 (as u32), hardware cvt if available ----
__device__ __forceinline__ unsigned pk2bf(float lo, float hi) {
#if __has_builtin(__builtin_amdgcn_cvt_pk_bf16_f32)
    v2bf p = __builtin_amdgcn_cvt_pk_bf16_f32(lo, hi);
    return __builtin_bit_cast(unsigned, p);
#elif __has_builtin(__builtin_amdgcn_perm)
    // truncate-pack: result = {hi[31:16], lo[31:16]}
    return __builtin_amdgcn_perm(__builtin_bit_cast(unsigned, hi),
                                 __builtin_bit_cast(unsigned, lo), 0x07060302u);
#else
    unsigned ul = __builtin_bit_cast(unsigned, lo), uh = __builtin_bit_cast(unsigned, hi);
    ul = (ul + 0x7FFFu + ((ul >> 16) & 1u)) >> 16;
    uh = (uh + 0x7FFFu + ((uh >> 16) & 1u)) & 0xFFFF0000u;
    return uh | ul;
#endif
}

__device__ __forceinline__ float fast_exp2(float x) {
#if __has_builtin(__builtin_amdgcn_exp2f)
    return __builtin_amdgcn_exp2f(x);
#else
    return __expf(x * 0.6931471805599453f);
#endif
}

// 16-bit A-matrix 16x32 fragment (ISA 7.12.2): lane holds row = lane%16;
// elements 0..7  -> K = kchunk*32 + hi*8 + e        (contiguous 16B)
// elements 8..15 -> K = kchunk*32 + hi*8 + 16 + e   (contiguous 16B)
__device__ __forceinline__ v16bf load_frag_A(const __bf16* base, int row, int kchunk, int hi) {
    const __bf16* p = base + row * 64 + kchunk * 32 + hi * 8;
    v8bf lo = *(const v8bf*)(p);
    v8bf up = *(const v8bf*)(p + 16);
    v16bf r;
#pragma unroll
    for (int i = 0; i < 8; ++i) { r[i] = lo[i]; r[i + 8] = up[i]; }
    return r;
}

// 16-bit B-matrix 32x16 fragment: lane holds column N = lane%16 (= stored row);
// elements 0..7 -> K = kchunk*32 + hi*8 + e; elements 8..15 -> K = kchunk*32 + hi*8 + 16 + e
__device__ __forceinline__ v16bf load_frag_B(const __bf16* base, int row, int kchunk, int hi) {
    const __bf16* p = base + row * 64 + kchunk * 32 + hi * 8;
    v8bf lo = *(const v8bf*)(p);
    v8bf up = *(const v8bf*)(p + 16);
    v16bf r;
#pragma unroll
    for (int i = 0; i < 8; ++i) { r[i] = lo[i]; r[i + 8] = up[i]; }
    return r;
}

__global__ __launch_bounds__(NTHREADS) void attn_fwd(
    const float* __restrict__ Q, const float* __restrict__ K,
    const float* __restrict__ V, const unsigned char* __restrict__ Msk,
    const float* __restrict__ Bias, float* __restrict__ Out)
{
    __shared__ alignas(16) __bf16 sQ[BM * D_];              // [m][d] row-major
    __shared__ alignas(16) __bf16 sK[2][BN * D_];           // double-buffered [j][d]
    __shared__ alignas(16) __bf16 sVt[2][D_ * BN];          // double-buffered [d][j]
    __shared__ alignas(16) __bf16 sP[NWAVES][16 * BN];      // per-wave P strip

    const int tid  = threadIdx.x;
    const int lane = tid & 31;
    const int wave = tid >> 5;
    const int ln   = lane & 15;
    const int hi   = lane >> 4;

    // Grid order: b fastest (bias shared across b -> L2 catches x4 reuse),
    // h next (mask shared across h), q-tile slowest.
    const int b  = blockIdx.x & (B_ - 1);
    const int h  = (blockIdx.x >> 2) & (H_ - 1);
    const int qt = blockIdx.x >> 6;
    const int q0 = qt * BM;

    const float* qg = Q + (((size_t)(b * H_ + h) * N_) + q0) * D_;
    const float* kg = K + ((size_t)(b * H_ + h) * N_) * D_;
    const float* vg = V + ((size_t)(b * H_ + h) * N_) * D_;

    // ---- load + convert Q tile (128x64 fp32 -> bf16): 32 elems/thread ----
    for (int i = tid * 16; i < BM * D_; i += NTHREADS * 16) {
        float4 f[4];
#pragma unroll
        for (int u = 0; u < 4; ++u) f[u] = *(const float4*)(qg + i + u * 4);
        const float* fr = (const float*)&f[0];
        uint4 p0, p1;
        p0.x = pk2bf(fr[0],  fr[1]);  p0.y = pk2bf(fr[2],  fr[3]);
        p0.z = pk2bf(fr[4],  fr[5]);  p0.w = pk2bf(fr[6],  fr[7]);
        p1.x = pk2bf(fr[8],  fr[9]);  p1.y = pk2bf(fr[10], fr[11]);
        p1.z = pk2bf(fr[12], fr[13]); p1.w = pk2bf(fr[14], fr[15]);
        *(uint4*)(sQ + i) = p0;
        *(uint4*)(sQ + i + 8) = p1;
    }

    // ---- software-pipelined K/V tile fill ----
    const int jb = (tid >> 4) * 4;   // 0..60
    const int db = (tid & 15) * 4;   // 0..60
    float4 kreg[4], vreg[4];

    auto load_tile_regs = [&](int j0) {
        const float* kp = kg + (size_t)j0 * D_ + tid * 16;
#pragma unroll
        for (int u = 0; u < 4; ++u) kreg[u] = *(const float4*)(kp + u * 4);
        const float* vp = vg + ((size_t)j0 + jb) * D_ + db;
#pragma unroll
        for (int u = 0; u < 4; ++u) vreg[u] = *(const float4*)(vp + (size_t)u * D_);
    };
    auto store_tile_lds = [&](int buf) {
        const float* kr = (const float*)&kreg[0];
        uint4 p0, p1;
        p0.x = pk2bf(kr[0],  kr[1]);  p0.y = pk2bf(kr[2],  kr[3]);
        p0.z = pk2bf(kr[4],  kr[5]);  p0.w = pk2bf(kr[6],  kr[7]);
        p1.x = pk2bf(kr[8],  kr[9]);  p1.y = pk2bf(kr[10], kr[11]);
        p1.z = pk2bf(kr[12], kr[13]); p1.w = pk2bf(kr[14], kr[15]);
        *(uint4*)(&sK[buf][0] + tid * 16) = p0;
        *(uint4*)(&sK[buf][0] + tid * 16 + 8) = p1;
        const float* v0 = (const float*)&vreg[0];
        const float* v1 = (const float*)&vreg[1];
        const float* v2 = (const float*)&vreg[2];
        const float* v3 = (const float*)&vreg[3];
#pragma unroll
        for (int u = 0; u < 4; ++u) {   // u = d (column) within the 4x4 block
            uint2 t2;
            t2.x = pk2bf(v0[u], v1[u]);   // j = jb+0, jb+1
            t2.y = pk2bf(v2[u], v3[u]);   // j = jb+2, jb+3
            *(uint2*)(&sVt[buf][0] + (db + u) * BN + jb) = t2;
        }
    };

    load_tile_regs(0);
    __syncthreads();   // sQ visible

    const v16bf aq0 = load_frag_A(sQ, wave * 16 + ln, 0, hi);
    const v16bf aq1 = load_frag_A(sQ, wave * 16 + ln, 1, hi);

    // Constant "ones column" B-fragment: column 0 of a virtual V-extension is
    // all ones -> WMMA computes rowsum(P) into lane ln==0 of an extra C tile.
    v16bf blv;
    {
        const __bf16 ov = (ln == 0) ? __builtin_bit_cast(__bf16, (unsigned short)0x3F80)
                                    : __builtin_bit_cast(__bf16, (unsigned short)0x0000);
#pragma unroll
        for (int i = 0; i < 16; ++i) blv[i] = ov;
    }

    v8f o_acc[4];
#pragma unroll
    for (int t = 0; t < 4; ++t) { v8f z = {}; o_acc[t] = z; }
    v8f o_l = {};                 // rowsum(P) accumulator (col 0 of virtual tile)
    float m2_i[8];                // running max, base-2 (log2e-scaled) domain
#pragma unroll
    for (int r = 0; r < 8; ++r) m2_i[r] = -1e30f;

    for (int jt = 0; jt < NTILES; ++jt) {
        const int j0 = jt * BN;
        const int buf = jt & 1;
        // Double-buffered: storing buf is safe — any wave reaching here passed
        // barrier jt-1, which implies all waves finished compute jt-2 (last
        // reader of this buffer). One barrier per iteration.
        store_tile_lds(buf);
        __syncthreads();
        if (jt + 1 < NTILES) load_tile_regs(j0 + BN);   // overlaps with compute

        const __bf16* sKb  = &sK[buf][0];
        const __bf16* sVtb = &sVt[buf][0];

        // ---- S strip = Q_strip (16xD) * K_tile^T, 4 col-tiles ----
        v8f s[4];
#pragma unroll
        for (int t = 0; t < 4; ++t) {
            v16bf bk0 = load_frag_B(sKb, t * 16 + ln, 0, hi);
            v16bf bk1 = load_frag_B(sKb, t * 16 + ln, 1, hi);
            v8f acc = {};
            acc = __builtin_amdgcn_wmma_f32_16x16x32_bf16(false, aq0, false, bk0,
                                                          (short)0, acc, false, false);
            acc = __builtin_amdgcn_wmma_f32_16x16x32_bf16(false, aq1, false, bk1,
                                                          (short)0, acc, false, false);
            s[t] = acc;
        }

        // ---- scale + bias + additive mask (3 independent ops, no flag chains):
        //   addend = (m-1) & 0xF1490000  ->  0.0f (m=1)  or  -9.96e29f (m=0)
        //   sv += addend   (exact for m=1; absorbs sv for m=0)
        const int row0 = q0 + wave * 16 + 8 * hi;
        float mx[8];
#pragma unroll
        for (int r = 0; r < 8; ++r) {
            const int irow = row0 + r;
            const float* bp = Bias + ((size_t)h * N_ + irow) * N_ + j0;
            const unsigned char* mp = Msk + ((size_t)b * N_ + irow) * N_ + j0;
            float rm = -1e30f;
#pragma unroll
            for (int t = 0; t < 4; ++t) {
                const int jc = t * 16 + ln;
                float sv = s[t][r] * SCALEF + bp[jc];
                const unsigned am = ((unsigned)mp[jc] - 1u) & 0xF1490000u;
                sv += __builtin_bit_cast(float, am);
                s[t][r] = sv;
                rm = fmaxf(rm, sv);
            }
            mx[r] = rm;
        }
        // row-max across the 16 lanes of each half-wave
#pragma unroll
        for (int r = 0; r < 8; ++r) {
            float rm = mx[r];
            rm = fmaxf(rm, __shfl_xor(rm, 1));
            rm = fmaxf(rm, __shfl_xor(rm, 2));
            rm = fmaxf(rm, __shfl_xor(rm, 4));
            rm = fmaxf(rm, __shfl_xor(rm, 8));
            mx[r] = rm;
        }

        // ---- online softmax (base-2 domain; rowsum comes from WMMA) ----
        float alpha[8];
#pragma unroll
        for (int r = 0; r < 8; ++r) {
            const float m2new = fmaxf(m2_i[r], mx[r] * LOG2E);
            alpha[r] = fast_exp2(m2_i[r] - m2new);
            m2_i[r] = m2new;
#pragma unroll
            for (int t = 0; t < 4; ++t)
                s[t][r] = fast_exp2(__builtin_fmaf(s[t][r], LOG2E, -m2new));
        }
#pragma unroll
        for (int t = 0; t < 4; ++t)
#pragma unroll
            for (int r = 0; r < 8; ++r)
                o_acc[t][r] *= alpha[r];
#pragma unroll
        for (int r = 0; r < 8; ++r)
            o_l[r] *= alpha[r];

        // ---- P (C-layout) -> wave-private LDS -> A-fragments ----
        __bf16* sPw = &sP[wave][0];
#pragma unroll
        for (int t = 0; t < 4; ++t) {
#pragma unroll
            for (int r = 0; r < 8; r += 2) {
                unsigned pk = pk2bf(s[t][r], s[t][r + 1]);   // one cvt for two rows
                const int base = (r + 8 * hi) * BN + t * 16 + ln;
                sPw[base]      = __builtin_bit_cast(__bf16, (unsigned short)(pk & 0xFFFFu));
                sPw[base + BN] = __builtin_bit_cast(__bf16, (unsigned short)(pk >> 16));
            }
        }
        v16bf ap0 = load_frag_A(sPw, ln, 0, hi);
        v16bf ap1 = load_frag_A(sPw, ln, 1, hi);

        // ---- O += P * V_tile (4 d-tiles) ; l += P * ones (constant B frag) ----
#pragma unroll
        for (int t = 0; t < 4; ++t) {
            v16bf bv0 = load_frag_B(sVtb, t * 16 + ln, 0, hi);
            v16bf bv1 = load_frag_B(sVtb, t * 16 + ln, 1, hi);
            o_acc[t] = __builtin_amdgcn_wmma_f32_16x16x32_bf16(false, ap0, false, bv0,
                                                               (short)0, o_acc[t], false, false);
            o_acc[t] = __builtin_amdgcn_wmma_f32_16x16x32_bf16(false, ap1, false, bv1,
                                                               (short)0, o_acc[t], false, false);
        }
        o_l = __builtin_amdgcn_wmma_f32_16x16x32_bf16(false, ap0, false, blv,
                                                      (short)0, o_l, false, false);
        o_l = __builtin_amdgcn_wmma_f32_16x16x32_bf16(false, ap1, false, blv,
                                                      (short)0, o_l, false, false);
    }

    // ---- epilogue: O / l (l lives in lanes ln==0; broadcast within halves) ----
    float* og = Out + (((size_t)(b * H_ + h) * N_) + q0) * D_;
#pragma unroll
    for (int r = 0; r < 8; ++r) {
        const int mrow = wave * 16 + 8 * hi + r;
        const float lsum = __shfl(o_l[r], hi << 4);   // lane 0 / lane 16
        const float inv = 1.0f / lsum;
#pragma unroll
        for (int t = 0; t < 4; ++t)
            og[(size_t)mrow * D_ + t * 16 + ln] = o_acc[t][r] * inv;
    }
}

extern "C" void kernel_launch(void* const* d_in, const int* in_sizes, int n_in,
                              void* d_out, int out_size, void* d_ws, size_t ws_size,
                              hipStream_t stream) {
    (void)in_sizes; (void)n_in; (void)out_size; (void)d_ws; (void)ws_size;
    const float* q = (const float*)d_in[0];
    const float* k = (const float*)d_in[1];
    const float* v = (const float*)d_in[2];
    const unsigned char* mask = (const unsigned char*)d_in[3];
    const float* bias = (const float*)d_in[4];
    float* out = (float*)d_out;

    dim3 grid(B_ * H_ * (N_ / BM));   // 1024, b fastest
    attn_fwd<<<grid, NTHREADS, 0, stream>>>(q, k, v, mask, bias, out);
}
